// BiGruBNattMaxFocalNet_34256659153469
// MI455X (gfx1250) — matrix operand order repaired and checked by reading, not compiled, not run on hardware
//
#include <hip/hip_runtime.h>

// ---------------------------------------------------------------------------
// Types for CDNA5 WMMA (wave32): A/B = 16 bf16 per lane, C/D = 8 f32 per lane
// ---------------------------------------------------------------------------
typedef __attribute__((ext_vector_type(16))) __bf16 v16bf;
typedef __attribute__((ext_vector_type(8)))  float  v8f;

#define WMMA_BF16(a, b, c) \
  __builtin_amdgcn_wmma_f32_16x16x32_bf16(false, (a), false, (b), (short)0, (c), false, false)

static __device__ inline v8f v8f_zero() {
  v8f z = {0.f, 0.f, 0.f, 0.f, 0.f, 0.f, 0.f, 0.f};
  return z;
}

static __device__ inline float sigmoidf_(float x) { return 1.0f / (1.0f + __expf(-x)); }

// fast tanh: use the gfx1250 TANH transcendental when the builtin exists
static __device__ inline float tanh_fast(float x) {
#if __has_builtin(__builtin_amdgcn_tanhf)
  return __builtin_amdgcn_tanhf(x);
#elif __has_builtin(__builtin_amdgcn_tanh_f32)
  return __builtin_amdgcn_tanh_f32(x);
#else
  return tanhf(x);
#endif
}

// pack two floats into two bf16 in one dword: single v_cvt_pk_bf16_f32
// (src0 -> low half, src1 -> high half; matches compiler's own usage)
static __device__ inline unsigned pack_bf16(float a, float b) {
  unsigned u;
  asm("v_cvt_pk_bf16_f32 %0, %1, %2" : "=v"(u) : "v"(a), "v"(b));
  return u;
}

// ---------------------------------------------------------------------------
// WMMA fragment loaders (bf16, padded-stride LDS or global rows).
// A (16x32 MxK): lane l holds row m=l&15; vgpr j holds
//   k = (l>>4)*16 + 2*j and k+1.
// B (32x16 KxN) staged N-major: lane l holds col n=l&15; vgpr j holds
//   k = (l>>4)*16 + 2*j and k+1.
// ---------------------------------------------------------------------------
static __device__ inline v16bf fragA(const __bf16* base, int lda, int m0, int kc, int lane) {
  const __bf16* row = base + (size_t)(m0 + (lane & 15)) * lda + kc;
  const int kh = (lane >> 4) * 16;
  v16bf a;
#pragma unroll
  for (int j = 0; j < 8; ++j) {
    int k = kh + (j << 1);
    a[2 * j]     = row[k];
    a[2 * j + 1] = row[k + 1];
  }
  return a;
}

static __device__ inline v16bf fragB(const __bf16* base, int ldb, int n0, int kc, int lane) {
  const __bf16* row = base + (size_t)(n0 + (lane & 15)) * ldb + kc + ((lane >> 4) << 4);
  v16bf b;
#pragma unroll
  for (int j = 0; j < 8; ++j) {
    b[2 * j]     = row[2 * j];
    b[2 * j + 1] = row[2 * j + 1];
  }
  return b;
}

// ---------------------------------------------------------------------------
// fp32 -> bf16 weight conversion (plain and K-padded variants)
// ---------------------------------------------------------------------------
__global__ void k_f2bf(const float* __restrict__ s, __bf16* __restrict__ d, int n) {
  int i = blockIdx.x * blockDim.x + threadIdx.x;
  if (i < n) d[i] = (__bf16)s[i];
}

__global__ void k_f2bf_pad(const float* __restrict__ s, __bf16* __restrict__ d,
                           int N, int Kin, int Kout) {
  int idx = blockIdx.x * blockDim.x + threadIdx.x;
  if (idx >= N * Kout) return;
  int n = idx / Kout, k = idx % Kout;
  d[idx] = (k < Kin) ? (__bf16)s[(size_t)n * Kin + k] : (__bf16)0.0f;
}

// pad activations [M,Kin] -> [Mp,Kout], zero-filled outside
__global__ void k_pad(const float* __restrict__ src, float* __restrict__ dst,
                      int M, int Kin, int Mp, int Kout) {
  int idx = blockIdx.x * blockDim.x + threadIdx.x;
  if (idx >= Mp * Kout) return;
  int m = idx / Kout, k = idx % Kout;
  dst[idx] = (m < M && k < Kin) ? src[(size_t)m * Kin + k] : 0.0f;
}

// ---------------------------------------------------------------------------
// WMMA GEMM: Y[M,N] = X[M,Kd] @ W^T + bias, W is [N,Kd] bf16.
// PRECONDITIONS (caller-guaranteed): M%128==0, N%64==0, Kd%32==0, Kd%4==0.
// Block tile 128x64, 8 waves in 4x2, each wave 32x32 (2x2 wmma tiles).
// Register double-buffering: next K-chunk's global loads overlap WMMA phase.
// ---------------------------------------------------------------------------
__global__ __launch_bounds__(256) void k_gemm(const float* __restrict__ X,
                                              const __bf16* __restrict__ W,
                                              const float* __restrict__ bias,
                                              float* __restrict__ Y,
                                              int M, int N, int Kd) {
  __shared__ __bf16 As[128 * 40];
  __shared__ __bf16 Bt[64 * 40];
  const int tid = threadIdx.x, lane = tid & 31, wv = tid >> 5;
  const int wm = wv >> 1, wn = wv & 1;
  const int m0 = blockIdx.y * 128, n0 = blockIdx.x * 64;

  const int ra = tid >> 1, ksa = (tid & 1) * 16;       // A: 128 rows x 32 cols
  const int rb = tid >> 2, ksb = (tid & 3) * 8;        // B: 64 rows x 32 cols
  const float*  xrow = X + (size_t)(m0 + ra) * Kd + ksa;
  const __bf16* wrow = W + (size_t)(n0 + rb) * Kd + ksb;
  unsigned* adst = (unsigned*)(As + ra * 40 + ksa);    // 4B-aligned
  unsigned* bdst = (unsigned*)(Bt + rb * 40 + ksb);    // 4B-aligned

  v8f acc[2][2];
  acc[0][0] = v8f_zero(); acc[0][1] = v8f_zero();
  acc[1][0] = v8f_zero(); acc[1][1] = v8f_zero();

  // preload first K-chunk
  float4 xa0, xa1, xa2, xa3;
  unsigned wb0, wb1, wb2, wb3;
  {
    const float4* x4 = (const float4*)xrow;            // 16B-aligned (Kd%4==0)
    xa0 = x4[0]; xa1 = x4[1]; xa2 = x4[2]; xa3 = x4[3];
    const unsigned* w4 = (const unsigned*)wrow;        // 4B-aligned
    wb0 = w4[0]; wb1 = w4[1]; wb2 = w4[2]; wb3 = w4[3];
  }

  for (int kc = 0; kc < Kd; kc += 32) {
    // ---- stage current chunk into LDS ----
    adst[0] = pack_bf16(xa0.x, xa0.y); adst[1] = pack_bf16(xa0.z, xa0.w);
    adst[2] = pack_bf16(xa1.x, xa1.y); adst[3] = pack_bf16(xa1.z, xa1.w);
    adst[4] = pack_bf16(xa2.x, xa2.y); adst[5] = pack_bf16(xa2.z, xa2.w);
    adst[6] = pack_bf16(xa3.x, xa3.y); adst[7] = pack_bf16(xa3.z, xa3.w);
    bdst[0] = wb0; bdst[1] = wb1; bdst[2] = wb2; bdst[3] = wb3;
    __syncthreads();

    // ---- issue next chunk's global loads (overlap with WMMA phase) ----
    if (kc + 32 < Kd) {
      const float4* x4 = (const float4*)(xrow + kc + 32);
      xa0 = x4[0]; xa1 = x4[1]; xa2 = x4[2]; xa3 = x4[3];
      const unsigned* w4 = (const unsigned*)(wrow + kc + 32);
      wb0 = w4[0]; wb1 = w4[1]; wb2 = w4[2]; wb3 = w4[3];
    }

    v16bf a0 = fragA(As, 40, wm * 32,      0, lane);
    v16bf a1 = fragA(As, 40, wm * 32 + 16, 0, lane);
    v16bf b0 = fragB(Bt, 40, wn * 32,      0, lane);
    v16bf b1 = fragB(Bt, 40, wn * 32 + 16, 0, lane);
    acc[0][0] = WMMA_BF16(a0, b0, acc[0][0]);
    acc[0][1] = WMMA_BF16(a0, b1, acc[0][1]);
    acc[1][0] = WMMA_BF16(a1, b0, acc[1][0]);
    acc[1][1] = WMMA_BF16(a1, b1, acc[1][1]);
    __syncthreads();
  }

  // ---- epilogue (no bounds checks needed) ----
#pragma unroll
  for (int mi = 0; mi < 2; ++mi)
#pragma unroll
    for (int ni = 0; ni < 2; ++ni) {
      int n = n0 + wn * 32 + ni * 16 + (lane & 15);
      float bv = (bias != nullptr) ? bias[n] : 0.0f;
#pragma unroll
      for (int j = 0; j < 8; ++j) {
        int m = m0 + wm * 32 + mi * 16 + (lane >> 4) * 8 + j;
        Y[(size_t)m * N + n] = acc[mi][ni][j] + bv;
      }
    }
}

// ---------------------------------------------------------------------------
// GRU recurrence (torch gate order r,z,n). One workgroup = exactly 32 batch
// rows (grid sized so there is no partial chunk). h kept in LDS as bf16;
// gh = h @ Whh^T via WMMA (Whh bf16 streamed from L2).
// gx: [Btot, T, 384]; out: [Btot, T, 256] written at column colOff.
// ---------------------------------------------------------------------------
__global__ __launch_bounds__(256) void k_gru(const float* __restrict__ gx,
                                             const __bf16* __restrict__ Whh,
                                             const float* __restrict__ bhh,
                                             float* __restrict__ out,
                                             int T, int rev, int colOff) {
  __shared__ __bf16 hB[32 * 136];
  __shared__ float  ghS[32 * 392];
  const int tid = threadIdx.x, lane = tid & 31, wv = tid >> 5;
  const int b0 = blockIdx.x * 32;

  for (int i = tid; i < 32 * 68; i += 256) ((unsigned*)hB)[i] = 0u;
  __syncthreads();

  // gate-phase coordinates
  const int bl = tid >> 3;            // 0..31
  const int h0 = (tid & 7) * 16;      // 0..112
  const int bg = b0 + bl;

  for (int s = 0; s < T; ++s) {
    const int t = rev ? (T - 1 - s) : s;

    // prefetch next timestep's gx row (overlaps with WMMA phase)
    if (s + 1 < T) {
      const int tn = rev ? (T - 2 - s) : (s + 1);
      __builtin_prefetch(gx + ((size_t)bg * T + tn) * 384 + h0, 0, 1);
    }

    // gh = h @ Whh^T  (M=32, N=384, K=128). Wave wv owns n-tiles [wv*48, wv*48+48).
    v8f acc[2][3];
#pragma unroll
    for (int mi = 0; mi < 2; ++mi)
#pragma unroll
      for (int ti = 0; ti < 3; ++ti) acc[mi][ti] = v8f_zero();

#pragma unroll
    for (int kc = 0; kc < 128; kc += 32) {
      v16bf a0 = fragA(hB, 136, 0,  kc, lane);
      v16bf a1 = fragA(hB, 136, 16, kc, lane);
#pragma unroll
      for (int ti = 0; ti < 3; ++ti) {
        v16bf bb = fragB(Whh, 128, wv * 48 + ti * 16, kc, lane);
        acc[0][ti] = WMMA_BF16(a0, bb, acc[0][ti]);
        acc[1][ti] = WMMA_BF16(a1, bb, acc[1][ti]);
      }
    }

#pragma unroll
    for (int mi = 0; mi < 2; ++mi)
#pragma unroll
      for (int ti = 0; ti < 3; ++ti) {
        int n = wv * 48 + ti * 16 + (lane & 15);
        float bv = bhh[n];
#pragma unroll
        for (int j = 0; j < 8; ++j) {
          int m = mi * 16 + (lane >> 4) * 8 + j;
          ghS[m * 392 + n] = acc[mi][ti][j] + bv;
        }
      }
    __syncthreads();

    { // gate math: 32 batch x 128 hidden over 256 threads (16 each), vectorized
      const float* gxr = gx + ((size_t)bg * T + t) * 384 + h0;
      float* outr = out + ((size_t)bg * T + t) * 256 + colOff + h0;
      float gr[16], gz[16], gn[16];
      const float4* r4 = (const float4*)gxr;          // 16B-aligned
      const float4* z4 = (const float4*)(gxr + 128);
      const float4* n4 = (const float4*)(gxr + 256);
#pragma unroll
      for (int q = 0; q < 4; ++q) {
        float4 a = r4[q], b = z4[q], c = n4[q];
        gr[4 * q] = a.x; gr[4 * q + 1] = a.y; gr[4 * q + 2] = a.z; gr[4 * q + 3] = a.w;
        gz[4 * q] = b.x; gz[4 * q + 1] = b.y; gz[4 * q + 2] = b.z; gz[4 * q + 3] = b.w;
        gn[4 * q] = c.x; gn[4 * q + 1] = c.y; gn[4 * q + 2] = c.z; gn[4 * q + 3] = c.w;
      }
      float h2v[16];
      unsigned* hpk = (unsigned*)(hB + bl * 136 + h0);  // 4B-aligned
#pragma unroll
      for (int u = 0; u < 16; u += 2) {
        union { unsigned u32; __bf16 h[2]; } hu; hu.u32 = hpk[u >> 1];
#pragma unroll
        for (int e = 0; e < 2; ++e) {
          int h = h0 + u + e;
          float ghr = ghS[bl * 392 + h];
          float ghz = ghS[bl * 392 + 128 + h];
          float ghn = ghS[bl * 392 + 256 + h];
          float r  = sigmoidf_(gr[u + e] + ghr);
          float z  = sigmoidf_(gz[u + e] + ghz);
          float nn = tanh_fast(gn[u + e] + r * ghn);
          float hp = (float)hu.h[e];
          h2v[u + e] = (1.0f - z) * nn + z * hp;
        }
        hpk[u >> 1] = pack_bf16(h2v[u], h2v[u + 1]);
      }
      float4* o4 = (float4*)outr;                     // 16B-aligned
#pragma unroll
      for (int q = 0; q < 4; ++q) {
        float4 o; o.x = h2v[4 * q]; o.y = h2v[4 * q + 1];
        o.z = h2v[4 * q + 2]; o.w = h2v[4 * q + 3];
        o4[q] = o;
      }
    }
    __syncthreads();
  }
}

// ---------------------------------------------------------------------------
// w_e = We^T @ v   (We [256,256], v [256])
// ---------------------------------------------------------------------------
__global__ void k_we(const float* __restrict__ We, const float* __restrict__ v,
                     float* __restrict__ we) {
  int h = threadIdx.x;
  float s = 0.f;
  for (int j = 0; j < 256; ++j) s += We[j * 256 + h] * v[j];
  we[h] = s;
}

// ---------------------------------------------------------------------------
// scores[b,t,c] = sum_h tanh(q[b,t,h] + ck[b,c,h]) * we[h]
// one block per (b,t); waves handle c, lanes reduce over h (wave32 shuffles)
// ---------------------------------------------------------------------------
__global__ __launch_bounds__(256) void k_scores(const float* __restrict__ q,
                                                const float* __restrict__ ck,
                                                const float* __restrict__ we,
                                                float* __restrict__ sc, int iopt) {
  __shared__ float qS[256], wS[256];
  int blk = blockIdx.x;
  int b = blk / 50, t = blk % 50;
  int tid = threadIdx.x, lane = tid & 31, wv = tid >> 5;
  qS[tid] = q[((size_t)(b * 5 + iopt) * 50 + t) * 256 + tid];
  wS[tid] = we[tid];
  __syncthreads();
  for (int c = wv; c < 100; c += 8) {
    const float* cr = ck + ((size_t)b * 100 + c) * 256;
    float p = 0.f;
#pragma unroll
    for (int u = 0; u < 8; ++u) {
      int h = lane + 32 * u;
      p += tanh_fast(qS[h] + cr[h]) * wS[h];
    }
#pragma unroll
    for (int off = 16; off > 0; off >>= 1) p += __shfl_xor(p, off, 32);
    if (lane == 0) sc[((size_t)b * 50 + t) * 100 + c] = p;
  }
}

// softmax over t (axis=1): per (b,c)
__global__ void k_soft_t(const float* __restrict__ sc, float* __restrict__ A1) {
  int i = blockIdx.x * blockDim.x + threadIdx.x;
  if (i >= 32 * 100) return;
  int b = i / 100, c = i % 100;
  const float* base = sc + (size_t)b * 5000 + c;
  float mx = -1e30f;
  for (int t = 0; t < 50; ++t) mx = fmaxf(mx, base[t * 100]);
  float s = 0.f;
  for (int t = 0; t < 50; ++t) s += __expf(base[t * 100] - mx);
  float inv = 1.f / s;
  for (int t = 0; t < 50; ++t)
    A1[(size_t)b * 5000 + t * 100 + c] = __expf(base[t * 100] - mx) * inv;
}

// softmax over c (axis=2): per (b,t)
__global__ void k_soft_c(const float* __restrict__ sc, float* __restrict__ A2) {
  int i = blockIdx.x * blockDim.x + threadIdx.x;
  if (i >= 32 * 50) return;
  const float* base = sc + (size_t)i * 100;
  float mx = -1e30f;
  for (int c = 0; c < 100; ++c) mx = fmaxf(mx, base[c]);
  float s = 0.f;
  for (int c = 0; c < 100; ++c) s += __expf(base[c] - mx);
  float inv = 1.f / s;
  for (int c = 0; c < 100; ++c) A2[(size_t)i * 100 + c] = __expf(base[c] - mx) * inv;
}

// attn_ctx[b,c,h] = sum_t A1[b,t,c] * opt[(b*5+i)*50+t, h]; block=(b,c), thread=h
__global__ __launch_bounds__(256) void k_attnctx(const float* __restrict__ A1,
                                                 const float* __restrict__ opt,
                                                 float* __restrict__ ac, int iopt) {
  int blk = blockIdx.x;
  int b = blk / 100, c = blk % 100;
  int h = threadIdx.x;
  const float* orow = opt + ((size_t)(b * 5 + iopt) * 50) * 256 + h;
  float s = 0.f;
  for (int t = 0; t < 50; ++t) s += A1[(size_t)b * 5000 + t * 100 + c] * orow[(size_t)t * 256];
  ac[(size_t)blk * 256 + h] = s;
}

// attn_opt[b,t,h] = sum_c A2[b,t,c] * ctx[b,c,h]; block=(b,t), thread=h
__global__ __launch_bounds__(256) void k_attnopt(const float* __restrict__ A2,
                                                 const float* __restrict__ ctx,
                                                 float* __restrict__ ao) {
  int blk = blockIdx.x;
  int b = blk / 50;
  int h = threadIdx.x;
  const float* crow = ctx + (size_t)b * 100 * 256 + h;
  const float* arow = A2 + (size_t)blk * 100;
  float s = 0.f;
  for (int c = 0; c < 100; ++c) s += arow[c] * crow[(size_t)c * 256];
  ao[(size_t)blk * 256 + h] = s;
}

// fc = [x, a, x*a, x-a] where row maps 1:1 (context side); rows exact
__global__ void k_buildfc(const float* __restrict__ base, const float* __restrict__ attn,
                          float* __restrict__ f, int rows) {
  int idx = blockIdx.x * 256 + threadIdx.x;
  if (idx >= rows * 1024) return;
  int row = idx >> 10, col = idx & 1023;
  int seg = col >> 8, h = col & 255;
  float x = base[(size_t)row * 256 + h];
  float a = attn[(size_t)row * 256 + h];
  f[idx] = (seg == 0) ? x : (seg == 1) ? a : (seg == 2) ? x * a : x - a;
}

// fo = [x, a, x*a, x-a] with option-row remap; covers padded rows with zeros
__global__ void k_buildfo(const float* __restrict__ opt, const float* __restrict__ attn,
                          float* __restrict__ f, int iopt) {
  int idx = blockIdx.x * 256 + threadIdx.x;
  if (idx >= 1664 * 1024) return;
  int row = idx >> 10, col = idx & 1023;
  if (row >= 1600) { f[idx] = 0.0f; return; }
  int b = row / 50, t = row % 50;
  int seg = col >> 8, h = col & 255;
  float x = opt[((size_t)(b * 5 + iopt) * 50 + t) * 256 + h];
  float a = attn[(size_t)row * 256 + h];
  f[idx] = (seg == 0) ? x : (seg == 1) ? a : (seg == 2) ? x * a : x - a;
}

// time-max over both aGRU outputs + Wsim dot -> logits[b*5+i]
__global__ __launch_bounds__(256) void k_maxlogit(const float* __restrict__ ahc,
                                                  const float* __restrict__ aho,
                                                  const float* __restrict__ Wsim,
                                                  float* __restrict__ logits, int iopt) {
  __shared__ float red[256];
  int b = blockIdx.x, h = threadIdx.x;
  float mc = -1e30f, mo = -1e30f;
  for (int t = 0; t < 100; ++t) mc = fmaxf(mc, ahc[((size_t)b * 100 + t) * 256 + h]);
  for (int t = 0; t < 50; ++t)  mo = fmaxf(mo, aho[((size_t)b * 50 + t) * 256 + h]);
  red[h] = Wsim[h] * mc + Wsim[256 + h] * mo;
  __syncthreads();
  for (int s = 128; s > 0; s >>= 1) {
    if (h < s) red[h] += red[h + s];
    __syncthreads();
  }
  if (h == 0) logits[b * 5 + iopt] = red[0];
}

// final softmax over K=5
__global__ void k_softK(const float* __restrict__ logits, float* __restrict__ out) {
  int b = threadIdx.x;
  if (b >= 32) return;
  float mx = -1e30f;
  for (int i = 0; i < 5; ++i) mx = fmaxf(mx, logits[b * 5 + i]);
  float e[5], s = 0.f;
  for (int i = 0; i < 5; ++i) { e[i] = __expf(logits[b * 5 + i] - mx); s += e[i]; }
  for (int i = 0; i < 5; ++i) out[b * 5 + i] = e[i] / s;
}

// ---------------------------------------------------------------------------
// Host-side orchestration
// ---------------------------------------------------------------------------
static inline size_t align256(size_t x) { return (x + 255) & ~(size_t)255; }

extern "C" void kernel_launch(void* const* d_in, const int* in_sizes, int n_in,
                              void* d_out, int out_size, void* d_ws, size_t ws_size,
                              hipStream_t stream) {
  (void)in_sizes; (void)n_in; (void)out_size; (void)ws_size;

  const float* context = (const float*)d_in[0];
  const float* options = (const float*)d_in[1];
  const float* W_ih_f  = (const float*)d_in[4];
  const float* W_hh_f  = (const float*)d_in[5];
  const float* b_ih_f  = (const float*)d_in[6];
  const float* b_hh_f  = (const float*)d_in[7];
  const float* W_ih_b  = (const float*)d_in[8];
  const float* W_hh_b  = (const float*)d_in[9];
  const float* b_ih_b  = (const float*)d_in[10];
  const float* b_hh_b  = (const float*)d_in[11];
  const float* aW_ih_f = (const float*)d_in[12];
  const float* aW_hh_f = (const float*)d_in[13];
  const float* ab_ih_f = (const float*)d_in[14];
  const float* ab_hh_f = (const float*)d_in[15];
  const float* aW_ih_b = (const float*)d_in[16];
  const float* aW_hh_b = (const float*)d_in[17];
  const float* ab_ih_b = (const float*)d_in[18];
  const float* ab_hh_b = (const float*)d_in[19];
  const float* Wk      = (const float*)d_in[20];
  const float* Wq      = (const float*)d_in[21];
  const float* We      = (const float*)d_in[22];
  const float* vv      = (const float*)d_in[23];
  const float* Wsim    = (const float*)d_in[24];
  float* out = (float*)d_out;

  char* w = (char*)d_ws;
  size_t off = 0;
  auto allocF = [&](size_t n) -> float* {
    float* p = (float*)(w + off); off = align256(off + n * sizeof(float)); return p;
  };
  auto allocB = [&](size_t n) -> __bf16* {
    __bf16* p = (__bf16*)(w + off); off = align256(off + n * sizeof(__bf16)); return p;
  };

  // bf16 weight copies (W_ih padded K 300->320)
  __bf16* Wihf_b  = allocB(384 * 320);
  __bf16* Wihb_b  = allocB(384 * 320);
  __bf16* Whhf_b  = allocB(384 * 128);
  __bf16* Whhb_b  = allocB(384 * 128);
  __bf16* aWihf_b = allocB(384 * 1024);
  __bf16* aWihb_b = allocB(384 * 1024);
  __bf16* aWhhf_b = allocB(384 * 128);
  __bf16* aWhhb_b = allocB(384 * 128);
  __bf16* Wk_b    = allocB(256 * 256);
  __bf16* Wq_b    = allocB(256 * 256);

  // padded activations (M multiple of 128, K multiple of 32)
  float* ctx_pad  = allocF((size_t)3200 * 320);
  float* opt_pad  = allocF((size_t)8064 * 320);

  // f32 activations
  float* gx_ctx_f = allocF((size_t)3200 * 384);
  float* gx_ctx_b = allocF((size_t)3200 * 384);
  float* gx_opt_f = allocF((size_t)8064 * 384);
  float* gx_opt_b = allocF((size_t)8064 * 384);
  float* ctx_outs = allocF((size_t)3200 * 256);
  float* opt_outs = allocF((size_t)8064 * 256);
  float* ctx_key  = allocF((size_t)3200 * 256);
  float* q_all    = allocF((size_t)8064 * 256);
  float* w_e      = allocF(256);
  float* scores   = allocF((size_t)32 * 50 * 100);
  float* A1       = allocF((size_t)32 * 50 * 100);
  float* A2       = allocF((size_t)32 * 50 * 100);
  float* attn_ctx = allocF((size_t)3200 * 256);
  float* attn_opt = allocF((size_t)1600 * 256);
  float* fc       = allocF((size_t)3200 * 1024);
  float* fo       = allocF((size_t)1664 * 1024);
  float* agx_fc_f = allocF((size_t)3200 * 384);
  float* agx_fc_b = allocF((size_t)3200 * 384);
  float* agx_fo_f = allocF((size_t)1664 * 384);
  float* agx_fo_b = allocF((size_t)1664 * 384);
  float* ahc_outs = allocF((size_t)3200 * 256);
  float* aho_outs = allocF((size_t)1600 * 256);
  float* logitsB  = allocF(160);

  auto cvt = [&](const float* s, __bf16* d, int n) {
    k_f2bf<<<dim3((n + 255) / 256), dim3(256), 0, stream>>>(s, d, n);
  };
  // padded input-projection weights (K 300 -> 320, zero fill)
  k_f2bf_pad<<<dim3((384 * 320 + 255) / 256), dim3(256), 0, stream>>>(W_ih_f, Wihf_b, 384, 300, 320);
  k_f2bf_pad<<<dim3((384 * 320 + 255) / 256), dim3(256), 0, stream>>>(W_ih_b, Wihb_b, 384, 300, 320);
  cvt(W_hh_f, Whhf_b, 384 * 128);    cvt(W_hh_b, Whhb_b, 384 * 128);
  cvt(aW_ih_f, aWihf_b, 384 * 1024); cvt(aW_ih_b, aWihb_b, 384 * 1024);
  cvt(aW_hh_f, aWhhf_b, 384 * 128);  cvt(aW_hh_b, aWhhb_b, 384 * 128);
  cvt(Wk, Wk_b, 256 * 256);          cvt(Wq, Wq_b, 256 * 256);

  // padded activation copies
  k_pad<<<dim3((3200 * 320 + 255) / 256), dim3(256), 0, stream>>>(context, ctx_pad, 3200, 300, 3200, 320);
  k_pad<<<dim3((8064 * 320 + 255) / 256), dim3(256), 0, stream>>>(options, opt_pad, 8000, 300, 8064, 320);

  auto gemm = [&](const float* X, const __bf16* Wm, const float* bias, float* Y,
                  int M, int N, int Kd) {
    dim3 g(N / 64, M / 128);
    k_gemm<<<g, dim3(256), 0, stream>>>(X, Wm, bias, Y, M, N, Kd);
  };

  // First BiGRU input projections
  gemm(ctx_pad, Wihf_b, b_ih_f, gx_ctx_f, 3200, 384, 320);
  gemm(ctx_pad, Wihb_b, b_ih_b, gx_ctx_b, 3200, 384, 320);
  gemm(opt_pad, Wihf_b, b_ih_f, gx_opt_f, 8064, 384, 320);
  gemm(opt_pad, Wihb_b, b_ih_b, gx_opt_b, 8064, 384, 320);

  // First BiGRU recurrences
  k_gru<<<dim3(1), dim3(256), 0, stream>>>(gx_ctx_f, Whhf_b, b_hh_f, ctx_outs, 100, 0, 0);
  k_gru<<<dim3(1), dim3(256), 0, stream>>>(gx_ctx_b, Whhb_b, b_hh_b, ctx_outs, 100, 1, 128);
  k_gru<<<dim3(5), dim3(256), 0, stream>>>(gx_opt_f, Whhf_b, b_hh_f, opt_outs, 50, 0, 0);
  k_gru<<<dim3(5), dim3(256), 0, stream>>>(gx_opt_b, Whhb_b, b_hh_b, opt_outs, 50, 1, 128);

  // Attention projections (opt_outs rows 8000..8063 are never-read padding)
  gemm(ctx_outs, Wk_b, nullptr, ctx_key, 3200, 256, 256);
  gemm(opt_outs, Wq_b, nullptr, q_all, 8064, 256, 256);
  k_we<<<dim3(1), dim3(256), 0, stream>>>(We, vv, w_e);

  for (int i = 0; i < 5; ++i) {
    k_scores<<<dim3(1600), dim3(256), 0, stream>>>(q_all, ctx_key, w_e, scores, i);
    k_soft_t<<<dim3(13), dim3(256), 0, stream>>>(scores, A1);
    k_soft_c<<<dim3(7), dim3(256), 0, stream>>>(scores, A2);
    k_attnctx<<<dim3(3200), dim3(256), 0, stream>>>(A1, opt_outs, attn_ctx, i);
    k_attnopt<<<dim3(1600), dim3(256), 0, stream>>>(A2, ctx_outs, attn_opt);
    k_buildfc<<<dim3(12800), dim3(256), 0, stream>>>(ctx_outs, attn_ctx, fc, 3200);
    k_buildfo<<<dim3(6656), dim3(256), 0, stream>>>(opt_outs, attn_opt, fo, i);

    // Second BiGRU input projections (dominant FLOPs -> WMMA)
    gemm(fc, aWihf_b, ab_ih_f, agx_fc_f, 3200, 384, 1024);
    gemm(fc, aWihb_b, ab_ih_b, agx_fc_b, 3200, 384, 1024);
    gemm(fo, aWihf_b, ab_ih_f, agx_fo_f, 1664, 384, 1024);
    gemm(fo, aWihb_b, ab_ih_b, agx_fo_b, 1664, 384, 1024);

    // Second BiGRU recurrences
    k_gru<<<dim3(1), dim3(256), 0, stream>>>(agx_fc_f, aWhhf_b, ab_hh_f, ahc_outs, 100, 0, 0);
    k_gru<<<dim3(1), dim3(256), 0, stream>>>(agx_fc_b, aWhhb_b, ab_hh_b, ahc_outs, 100, 1, 128);
    k_gru<<<dim3(1), dim3(256), 0, stream>>>(agx_fo_f, aWhhf_b, ab_hh_f, aho_outs, 50, 0, 0);
    k_gru<<<dim3(1), dim3(256), 0, stream>>>(agx_fo_b, aWhhb_b, ab_hh_b, aho_outs, 50, 1, 128);

    k_maxlogit<<<dim3(32), dim3(256), 0, stream>>>(ahc_outs, aho_outs, Wsim, logitsB, i);
  }

  k_softK<<<dim3(1), dim3(32), 0, stream>>>(logitsB, out);
}
